// Model_39676907883409
// MI455X (gfx1250) — compile-verified, matrix operand order checked by reading, not run on hardware
//
#include <hip/hip_runtime.h>

typedef __attribute__((ext_vector_type(16))) _Float16 v16h;
typedef __attribute__((ext_vector_type(8)))  _Float16 v8h;
typedef __attribute__((ext_vector_type(4)))  _Float16 v4h;
typedef __attribute__((ext_vector_type(8)))  float    v8f;

#define HIDDEN   1024
#define NTOK     16          // tokens per workgroup tile
#define XSH      1064        // LDS row stride for f16 x tiles (halfs): 2128B rows -> bank-spread, 16B aligned
#define QS       3080        // LDS row stride for f32 qkv tile (dwords)
#define MATHI    (2u * 1024u * 1024u)   // bytes of packed hi-f16 per matrix
#define MATBLK   (4u * 1024u * 1024u)   // hi + lo per matrix

// ---------------------------------------------------------------------------
// Kernel 0: pack Wq/Wk/Wv (f32, row n = output feature, 1024 K each) into
// (hi,lo) f16 pairs laid out in exact WMMA-B fragment order:
//   per (ntile t, k0-group kk): 1KB block = [512B: elements 0-7, lane-major 16B]
//                                           [512B: elements 8-15, lane-major 16B]
// so the GEMM's B loads are coalesced global_load_b128.
// ---------------------------------------------------------------------------
__global__ __launch_bounds__(256) void pack_w(
    const float* __restrict__ Wq, const float* __restrict__ Wk,
    const float* __restrict__ Wv, char* __restrict__ ws)
{
    int gid = blockIdx.x * 256 + threadIdx.x;      // 3*1024*128 chunks of 8 K-values
    int mat = gid >> 17;                           // / (1024*128)
    int r   = gid & (1024 * 128 - 1);
    int n   = r >> 7;                              // weight row (output feature)
    int c   = r & 127;                             // 8-wide K chunk index
    const float* W = (mat == 0) ? Wq : ((mat == 1) ? Wk : Wv);
    const float* s = W + (size_t)n * HIDDEN + c * 8;

    int t      = n >> 4;                           // 16-wide N tile
    int l16    = n & 15;
    int kk     = c >> 2;                           // K=32 group
    int klocal = (c & 3) * 8;                      // 0,8,16,24 within the group
    int lane   = l16 + (((klocal >> 3) & 1) << 4); // K +8 -> upper lane half
    int selhi  = (klocal >= 16) ? 1 : 0;           // K +16 -> fragment elements 8-15

    size_t off = (size_t)mat * MATBLK
               + ((size_t)(t * 32 + kk) * 1024) + selhi * 512 + lane * 16;

    v8h h, l;
    #pragma unroll
    for (int j = 0; j < 8; ++j) {
        float v = s[j];
        _Float16 hv = (_Float16)v;
        h[j] = hv;
        l[j] = (_Float16)(v - (float)hv);
    }
    *(v8h*)(ws + off)         = h;                 // hi region
    *(v8h*)(ws + off + MATHI) = l;                 // lo region (+2MB)
}

// ---------------------------------------------------------------------------
// Kernel 1: fused QKV projection (f16 hi/lo split WMMA, f32 accumulate)
//           + per-token head-dim attention (reference quirk, negated scale)
// ---------------------------------------------------------------------------
__global__ __launch_bounds__(256) void fused_qkv_headattn(
    const float* __restrict__ x,
    const char*  __restrict__ wpk,   // packed weights (hi/lo f16, fragment order)
    const float* __restrict__ bq, const float* __restrict__ bk,
    const float* __restrict__ bv,
    float* __restrict__ out)
{
    extern __shared__ char smem[];
    _Float16* sxh  = (_Float16*)smem;                          // [NTOK][XSH] hi
    _Float16* sxl  = sxh + NTOK * XSH;                         // [NTOK][XSH] lo
    float*    sqkv = (float*)(smem + 2 * NTOK * XSH * 2);      // [NTOK][QS]

    const int tid   = threadIdx.x;
    const long tok0 = (long)blockIdx.x * NTOK;

    // ---- Phase 1: stage contiguous 16x1024 x-tile, split into f16 hi/lo ----
    {
        const float4* src = (const float4*)(x + tok0 * HIDDEN);
        #pragma unroll
        for (int i = 0; i < 16; ++i) {
            int idx = tid + 256 * i;              // 4096 float4 in tile
            float4 vv = src[idx];
            int row = idx >> 8;
            int col = (idx & 255) << 2;
            v4h h, l;
            const float* pv = (const float*)&vv;
            #pragma unroll
            for (int u = 0; u < 4; ++u) {
                _Float16 hv = (_Float16)pv[u];
                h[u] = hv;
                l[u] = (_Float16)(pv[u] - (float)hv);
            }
            *(v4h*)(&sxh[row * XSH + col]) = h;
            *(v4h*)(&sxl[row * XSH + col]) = l;
        }
    }
    __syncthreads();

    // ---- Phase 2: [16 x 3072] = xTile @ [Wq;Wk;Wv]^T + bias --------------
    {
        const int wave = tid >> 5;
        const int lane = tid & 31;
        const int half = lane >> 4;
        const int l16  = lane & 15;

        // A fragment bases: elements 0-7 at K = k0 + half*8, elements 8-15 at +16
        const _Float16* ahB = &sxh[l16 * XSH + half * 8];
        const _Float16* alB = &sxl[l16 * XSH + half * 8];

        union U16 { v16h v; struct { v8h lo, hi; } s; };

        #pragma unroll 1
        for (int nt = 0; nt < 24; ++nt) {
            int n0  = wave * 384 + nt * 16;       // 16-wide tiles never straddle a matrix
            int mat = n0 >> 10;
            int e0  = n0 & 1023;
            const float* bias = (mat == 0) ? bq : ((mat == 1) ? bk : bv);
            int tLocal = e0 >> 4;
            const char* bBase = wpk + (size_t)mat * MATBLK
                              + (size_t)tLocal * 32 * 1024 + lane * 16;

            v8f c = {0.f, 0.f, 0.f, 0.f, 0.f, 0.f, 0.f, 0.f};
            #pragma unroll 2
            for (int kk = 0; kk < 32; ++kk) {     // K = kk*32
                U16 ah, al, bh, bl;
                int k0 = kk * 32;
                ah.s.lo = *(const v8h*)(ahB + k0);
                ah.s.hi = *(const v8h*)(ahB + k0 + 16);
                al.s.lo = *(const v8h*)(alB + k0);
                al.s.hi = *(const v8h*)(alB + k0 + 16);
                const char* bp = bBase + (size_t)kk * 1024;
                bh.s.lo = *(const v8h*)(bp);
                bh.s.hi = *(const v8h*)(bp + 512);
                bl.s.lo = *(const v8h*)(bp + MATHI);
                bl.s.hi = *(const v8h*)(bp + MATHI + 512);
                // x*w ~= ah*bh + ah*bl + al*bh  (f32 accumulate)
                c = __builtin_amdgcn_wmma_f32_16x16x32_f16(false, ah.v, false, bh.v,
                                                           (short)0, c, false, false);
                c = __builtin_amdgcn_wmma_f32_16x16x32_f16(false, ah.v, false, bl.v,
                                                           (short)0, c, false, false);
                c = __builtin_amdgcn_wmma_f32_16x16x32_f16(false, al.v, false, bh.v,
                                                           (short)0, c, false, false);
            }
            float bb = bias[e0 + l16];
            #pragma unroll
            for (int j = 0; j < 8; ++j) {         // C: vgpr j -> row j + half*8, col l16
                int row = j + half * 8;
                sqkv[row * QS + n0 + l16] = c[j] + bb;
            }
        }
    }
    __syncthreads();

    // ---- Phase 3: per-token attention over the HEAD dim (8x8), neg scale ---
    {
        const int pair  = tid >> 1;               // 128 (token, head) pairs
        const int halfd = tid & 1;
        const int token = pair >> 3;
        const int head  = pair & 7;
        const float scale = -0.08838834764831845f;   // 1 / (-sqrt(128))

        const float* q     = &sqkv[token * QS + head * 128];
        const float* kbase = &sqkv[token * QS + 1024];
        const float* vbase = &sqkv[token * QS + 2048];

        float sc[8];
        #pragma unroll
        for (int g = 0; g < 8; ++g) {
            float s = 0.f;
            const float* kkp = kbase + g * 128;
            #pragma unroll 8
            for (int d = 0; d < 128; ++d) s = fmaf(q[d], kkp[d], s);
            sc[g] = s * scale;
        }
        float m = sc[0];
        #pragma unroll
        for (int g = 1; g < 8; ++g) m = fmaxf(m, sc[g]);
        float den = 0.f;
        #pragma unroll
        for (int g = 0; g < 8; ++g) { sc[g] = __expf(sc[g] - m); den += sc[g]; }
        float inv = 1.0f / den;

        float* op = out + (tok0 + token) * HIDDEN + head * 128 + halfd * 64;
        #pragma unroll
        for (int d = 0; d < 64; d += 4) {
            float4 o;
            float* oo = (float*)&o;
            #pragma unroll
            for (int u = 0; u < 4; ++u) {
                int dd = halfd * 64 + d + u;
                float acc = 0.f;
                #pragma unroll
                for (int g = 0; g < 8; ++g) acc = fmaf(sc[g], vbase[g * 128 + dd], acc);
                oo[u] = acc * inv;
            }
            *(float4*)(op + d) = o;
        }
    }
}

extern "C" void kernel_launch(void* const* d_in, const int* in_sizes, int n_in,
                              void* d_out, int out_size, void* d_ws, size_t ws_size,
                              hipStream_t stream) {
    (void)in_sizes; (void)n_in; (void)out_size; (void)ws_size;
    const float* x  = (const float*)d_in[0];
    const float* Wq = (const float*)d_in[1];
    const float* bq = (const float*)d_in[2];
    const float* Wk = (const float*)d_in[3];
    const float* bk = (const float*)d_in[4];
    const float* Wv = (const float*)d_in[5];
    const float* bv = (const float*)d_in[6];
    float* out = (float*)d_out;
    char* wpk  = (char*)d_ws;     // needs 12 MiB of scratch (3 mats x 4 MiB hi+lo)

    // Pack weights into f16 hi/lo WMMA-B fragment layout (runs each launch; tiny).
    pack_w<<<dim3(3 * 1024 * 128 / 256), dim3(256), 0, stream>>>(Wq, Wk, Wv, wpk);

    const size_t smemBytes = (size_t)(2 * NTOK * XSH * 2)       // hi+lo f16 x tiles
                           + (size_t)(NTOK * QS) * sizeof(float); // f32 qkv tile (~265 KB)
    hipFuncSetAttribute((const void*)fused_qkv_headattn,
                        hipFuncAttributeMaxDynamicSharedMemorySize, (int)smemBytes);

    dim3 grid(8 * 2048 / NTOK);   // 1024 token tiles
    dim3 block(256);              // 8 wave32
    fused_qkv_headattn<<<grid, block, smemBytes, stream>>>(x, wpk, bq, bk, bv, out);
}
